// ConvGRU_78194174591507
// MI455X (gfx1250) — matrix operand order repaired and checked by reading, not compiled, not run on hardware
//
#include <hip/hip_runtime.h>

typedef __attribute__((ext_vector_type(16))) __bf16 v16bf;
typedef __attribute__((ext_vector_type(8)))  float  v8f;

#define BD   8
#define HD   256
#define WD   256
#define CIN  16
#define CHN  32
#define CST  48          // stacked channels
#define CHPAD 72         // LDS per-pixel channel pad (144B col stride -> 16B aligned, conflict-free)
#define TILE_H 8
#define TILE_W 16
#define LROWS 10
#define LCOLS 18
#define NCHUNK 14        // ceil(9 taps * 3 groups / 2) K=32 chunks (dense K packing)
#define NFRAG  (3*2*NCHUNK)   // 84 fragments total
#define NW1    (2*2*NCHUNK)   // 56 fragments used by pass 1 (update+reset)
#define NW2    (1*2*NCHUNK)   // 28 fragments used by pass 2 (out)
#define LIN_OFF1 (NW1*512)    // input-tile offset (halves) in pass-1 dynamic LDS
#define LIN_OFF2 (NW2*512)

__device__ __forceinline__ unsigned short f2bf(float f) {
    union { float f; unsigned int u; } v; v.f = f;
    unsigned int u = v.u;
    u += 0x7FFFu + ((u >> 16) & 1u);          // round-to-nearest-even
    return (unsigned short)(u >> 16);
}
__device__ __forceinline__ float bf2f(unsigned short h) {
    union { unsigned int u; float f; } v; v.u = ((unsigned int)h) << 16;
    return v.f;
}

union FragU { uint4 q[2]; v16bf v; };

__device__ __forceinline__ v16bf load_frag(const unsigned short* p) {
    FragU u;
    const uint4* q = (const uint4*)p;         // 16B-aligned by construction
    u.q[0] = q[0]; u.q[1] = q[1];
    return u.v;
}

// Compile-time B-fragment tile offset for group g (halves).  Group g carries 16
// contiguous channels of one (tap, channel-range); g==27 is the zero pad
// (channels 48..63 of the tile are zeroed).
constexpr int bofs_c(int g) {
    int t  = (g >= 27) ? 8  : g / 3;
    int cr = (g >= 27) ? 48 : (g % 3) * 16;
    int dy = t / 3, dx = t % 3;
    return (dy * LCOLS + dx) * CHPAD + cr;
}

// ---------------------------------------------------------------------------
// Repack OIHW f32 filters into WMMA bf16 A-fragments with the dense-K chunk
// schedule.  frag f = (conv*2 + mtile)*14 + chunk ; conv: 0=update 1=reset 2=out
// A layout: lane L: M = L%16 ; element e: K = e + (L>>4)*16
// K slot -> group g = 2*chunk + (K>>4); channel = (g%3)*16 + (K&15); tap = g/3.
// ---------------------------------------------------------------------------
__global__ void pack_weights(const float* __restrict__ wu,
                             const float* __restrict__ wr,
                             const float* __restrict__ wo,
                             unsigned short* __restrict__ out) {
    int t = blockIdx.x * blockDim.x + threadIdx.x;
    if (t >= NFRAG * 32) return;
    int lane = t & 31, f = t >> 5;
    int chunk = f % NCHUNK;
    int mc    = f / NCHUNK;
    int mt    = mc & 1;
    int conv  = mc >> 1;
    const float* Wm = (conv == 0) ? wu : ((conv == 1) ? wr : wo);
    int M  = mt * 16 + (lane & 15);
    int hi = lane >> 4;
    unsigned short* dst = out + (size_t)f * 512 + lane * 16;
#pragma unroll
    for (int e = 0; e < 16; ++e) {
        int K = e + hi * 16;        // 0..31
        int g = 2 * chunk + (K >> 4);
        unsigned short v = 0;
        if (g < 27) {
            int tap = g / 3;
            int ch  = (g % 3) * 16 + (K & 15);
            v = f2bf(Wm[(M * CST + ch) * 9 + tap]);
        }
        dst[e] = v;
    }
}

// Cooperative copy of nfrag weight fragments into LDS (128-bit moves).
__device__ __forceinline__ void stage_weights(const unsigned short* __restrict__ src,
                                              unsigned short* dst, int nfrag) {
    const uint4* s = (const uint4*)src;
    uint4* d = (uint4*)dst;
    for (int i = threadIdx.x; i < nfrag * 64; i += 256) d[i] = s[i];
}

// Stage masked stacked input (bf16) with halo; channels 48..71 zeroed.
// chsel: 0 -> [x*m, prev*m] (pass 1);  1 -> [x*m, prev*m*reset] (pass 2).
__device__ __forceinline__ void stage_input(
    unsigned short* lin, const float* __restrict__ x, const float* __restrict__ prev,
    const int* __restrict__ mask, const unsigned short* __restrict__ ws_r,
    int b, int h0, int w0, int chsel)
{
    for (int pos = threadIdx.x; pos < LROWS * LCOLS; pos += 256) {
        int row = pos / LCOLS, col = pos % LCOLS;
        int r = h0 - 1 + row, c = w0 - 1 + col;
        bool in = (r >= 0) && (r < HD) && (c >= 0) && (c < WD);
        float mv = in ? (float)mask[((size_t)b * HD + r) * WD + c] : 0.f;
        unsigned short* dst = &lin[pos * CHPAD];
        if (mv != 0.f) {
            const float* xp = x + (((size_t)b * CIN) * HD + r) * WD + c;
#pragma unroll
            for (int ch = 0; ch < CIN; ++ch) dst[ch] = f2bf(xp[(size_t)ch * HD * WD]);
            size_t pbase = (((size_t)b * CHN) * HD + r) * WD + c;
            if (chsel == 0) {
#pragma unroll 8
                for (int ch = 0; ch < CHN; ++ch)
                    dst[CIN + ch] = f2bf(prev[pbase + (size_t)ch * HD * WD]);
            } else {
#pragma unroll 8
                for (int ch = 0; ch < CHN; ++ch) {
                    size_t idx = pbase + (size_t)ch * HD * WD;
                    dst[CIN + ch] = f2bf(prev[idx] * bf2f(ws_r[idx]));
                }
            }
        } else {
#pragma unroll 8
            for (int ch = 0; ch < CST; ++ch) dst[ch] = 0;
        }
#pragma unroll
        for (int ch = CST; ch < CHPAD; ++ch) dst[ch] = 0;
    }
}

// ---------------------------------------------------------------------------
// Pass 1: update & reset gates.  8 waves, tile 8 rows x 16 cols.
// Dynamic LDS: [NW1 weight frags | input tile]  (83264 B)
// ---------------------------------------------------------------------------
__global__ __launch_bounds__(256) void convgru_pass1(
    const float* __restrict__ x, const float* __restrict__ prev,
    const int* __restrict__ mask, const unsigned short* __restrict__ wsw,
    const float* __restrict__ b_update, const float* __restrict__ b_reset,
    unsigned short* __restrict__ ws_u, unsigned short* __restrict__ ws_r)
{
    extern __shared__ unsigned short smem[];
    unsigned short* lw  = smem;             // 56 frags * 1KB
    unsigned short* lin = smem + LIN_OFF1;  // 10*18*72 halves

    const int b  = blockIdx.z;
    const int h0 = blockIdx.y * TILE_H;
    const int w0 = blockIdx.x * TILE_W;

    stage_weights(wsw, lw, NW1);            // pass-1 frags are wsw[0..55]
    stage_input(lin, x, prev, mask, nullptr, b, h0, w0, 0);
    __syncthreads();

    const int lane = threadIdx.x & 31;
    const int wave = threadIdx.x >> 5;
    const int n16  = lane & 15;
    const int hi   = lane >> 4;
    const int h = h0 + wave;
    const int w = w0 + n16;
    const int lo  = lane * 16;
    const int pix = (wave * LCOLS + n16) * CHPAD;

    v8f au0 = {0.f,0.f,0.f,0.f,0.f,0.f,0.f,0.f};
    v8f au1 = au0, ar0 = au0, ar1 = au0;

    // Software-pipelined over 14 K-chunks; all fragments come from LDS.
    v16bf bcur = load_frag(lin + pix + (hi ? bofs_c(1) : bofs_c(0)));
    v16bf acu0 = load_frag(lw + (0*NCHUNK + 0) * 512 + lo);
    v16bf acu1 = load_frag(lw + (1*NCHUNK + 0) * 512 + lo);
    v16bf acr0 = load_frag(lw + (2*NCHUNK + 0) * 512 + lo);
    v16bf acr1 = load_frag(lw + (3*NCHUNK + 0) * 512 + lo);

#pragma unroll
    for (int k = 0; k < NCHUNK; ++k) {
        v16bf bnxt, anu0, anu1, anr0, anr1;
        if (k + 1 < NCHUNK) {
            bnxt = load_frag(lin + pix + (hi ? bofs_c(2*(k+1)+1) : bofs_c(2*(k+1))));
            anu0 = load_frag(lw + (0*NCHUNK + k+1) * 512 + lo);
            anu1 = load_frag(lw + (1*NCHUNK + k+1) * 512 + lo);
            anr0 = load_frag(lw + (2*NCHUNK + k+1) * 512 + lo);
            anr1 = load_frag(lw + (3*NCHUNK + k+1) * 512 + lo);
        }
        au0 = __builtin_amdgcn_wmma_f32_16x16x32_bf16(false, acu0, false, bcur, (short)0, au0, false, false);
        au1 = __builtin_amdgcn_wmma_f32_16x16x32_bf16(false, acu1, false, bcur, (short)0, au1, false, false);
        ar0 = __builtin_amdgcn_wmma_f32_16x16x32_bf16(false, acr0, false, bcur, (short)0, ar0, false, false);
        ar1 = __builtin_amdgcn_wmma_f32_16x16x32_bf16(false, acr1, false, bcur, (short)0, ar1, false, false);
        bcur = bnxt; acu0 = anu0; acu1 = anu1; acr0 = anr0; acr1 = anr1;
    }

    const float mval = (float)mask[((size_t)b * HD + h) * WD + w];
    float bu0[8], bu1[8], br0[8], br1[8];
    *(float4*)&bu0[0] = ((const float4*)(b_update +      8*hi))[0];
    *(float4*)&bu0[4] = ((const float4*)(b_update +      8*hi))[1];
    *(float4*)&bu1[0] = ((const float4*)(b_update + 16 + 8*hi))[0];
    *(float4*)&bu1[4] = ((const float4*)(b_update + 16 + 8*hi))[1];
    *(float4*)&br0[0] = ((const float4*)(b_reset  +      8*hi))[0];
    *(float4*)&br0[4] = ((const float4*)(b_reset  +      8*hi))[1];
    *(float4*)&br1[0] = ((const float4*)(b_reset  + 16 + 8*hi))[0];
    *(float4*)&br1[4] = ((const float4*)(b_reset  + 16 + 8*hi))[1];
#pragma unroll
    for (int r = 0; r < 8; ++r) {
        int ch0 = r + 8 * hi;            // mtile 0
        int ch1 = 16 + r + 8 * hi;       // mtile 1
        size_t i0 = (((size_t)b * CHN + ch0) * HD + h) * WD + w;
        size_t i1 = (((size_t)b * CHN + ch1) * HD + h) * WD + w;
        float vu0 = mval / (1.f + __expf(-(au0[r] + bu0[r])));
        float vu1 = mval / (1.f + __expf(-(au1[r] + bu1[r])));
        float vr0 = mval / (1.f + __expf(-(ar0[r] + br0[r])));
        float vr1 = mval / (1.f + __expf(-(ar1[r] + br1[r])));
        ws_u[i0] = f2bf(vu0); ws_u[i1] = f2bf(vu1);
        ws_r[i0] = f2bf(vr0); ws_r[i1] = f2bf(vr1);
    }
}

// ---------------------------------------------------------------------------
// Pass 2: out-conv on [x*m, prev*m*reset], tanh, GRU blend -> d_out (f32).
// Dynamic LDS: [NW2 weight frags | input tile]  (54592 B)
// ---------------------------------------------------------------------------
__global__ __launch_bounds__(256) void convgru_pass2(
    const float* __restrict__ x, const float* __restrict__ prev,
    const int* __restrict__ mask, const unsigned short* __restrict__ wsw,
    const float* __restrict__ b_out,
    const unsigned short* __restrict__ ws_u, const unsigned short* __restrict__ ws_r,
    float* __restrict__ out)
{
    extern __shared__ unsigned short smem[];
    unsigned short* lw  = smem;             // 28 frags * 1KB
    unsigned short* lin = smem + LIN_OFF2;

    const int b  = blockIdx.z;
    const int h0 = blockIdx.y * TILE_H;
    const int w0 = blockIdx.x * TILE_W;

    stage_weights(wsw + (size_t)NW1 * 512, lw, NW2);   // pass-2 frags are wsw[56..83]
    stage_input(lin, x, prev, mask, ws_r, b, h0, w0, 1);
    __syncthreads();

    const int lane = threadIdx.x & 31;
    const int wave = threadIdx.x >> 5;
    const int n16  = lane & 15;
    const int hi   = lane >> 4;
    const int h = h0 + wave;
    const int w = w0 + n16;
    const int lo  = lane * 16;
    const int pix = (wave * LCOLS + n16) * CHPAD;

    v8f ao0 = {0.f,0.f,0.f,0.f,0.f,0.f,0.f,0.f};
    v8f ao1 = ao0;

    v16bf bcur = load_frag(lin + pix + (hi ? bofs_c(1) : bofs_c(0)));
    v16bf ac0  = load_frag(lw + (0*NCHUNK + 0) * 512 + lo);
    v16bf ac1  = load_frag(lw + (1*NCHUNK + 0) * 512 + lo);

#pragma unroll
    for (int k = 0; k < NCHUNK; ++k) {
        v16bf bnxt, an0, an1;
        if (k + 1 < NCHUNK) {
            bnxt = load_frag(lin + pix + (hi ? bofs_c(2*(k+1)+1) : bofs_c(2*(k+1))));
            an0  = load_frag(lw + (0*NCHUNK + k+1) * 512 + lo);
            an1  = load_frag(lw + (1*NCHUNK + k+1) * 512 + lo);
        }
        ao0 = __builtin_amdgcn_wmma_f32_16x16x32_bf16(false, ac0, false, bcur, (short)0, ao0, false, false);
        ao1 = __builtin_amdgcn_wmma_f32_16x16x32_bf16(false, ac1, false, bcur, (short)0, ao1, false, false);
        bcur = bnxt; ac0 = an0; ac1 = an1;
    }

    const float mval = (float)mask[((size_t)b * HD + h) * WD + w];
    float bo0[8], bo1[8];
    *(float4*)&bo0[0] = ((const float4*)(b_out +      8*hi))[0];
    *(float4*)&bo0[4] = ((const float4*)(b_out +      8*hi))[1];
    *(float4*)&bo1[0] = ((const float4*)(b_out + 16 + 8*hi))[0];
    *(float4*)&bo1[4] = ((const float4*)(b_out + 16 + 8*hi))[1];
#pragma unroll
    for (int r = 0; r < 8; ++r) {
        int ch0 = r + 8 * hi;
        int ch1 = 16 + r + 8 * hi;
        size_t i0 = (((size_t)b * CHN + ch0) * HD + h) * WD + w;
        size_t i1 = (((size_t)b * CHN + ch1) * HD + h) * WD + w;
        float o0 = tanhf(ao0[r] + bo0[r]) * mval;
        float o1 = tanhf(ao1[r] + bo1[r]) * mval;
        float u0 = bf2f(ws_u[i0]);
        float u1 = bf2f(ws_u[i1]);
        float p0 = prev[i0] * mval;
        float p1 = prev[i1] * mval;
        out[i0] = p0 * (1.f - u0) + o0 * u0;
        out[i1] = p1 * (1.f - u1) + o1 * u1;
    }
}

// ---------------------------------------------------------------------------
extern "C" void kernel_launch(void* const* d_in, const int* in_sizes, int n_in,
                              void* d_out, int out_size, void* d_ws, size_t ws_size,
                              hipStream_t stream) {
    const float* x        = (const float*)d_in[0];
    const float* prev     = (const float*)d_in[1];
    const int*   mask     = (const int*)d_in[2];
    const float* w_reset  = (const float*)d_in[3];
    const float* b_reset  = (const float*)d_in[4];
    const float* w_update = (const float*)d_in[5];
    const float* b_update = (const float*)d_in[6];
    const float* w_out    = (const float*)d_in[7];
    const float* b_out    = (const float*)d_in[8];
    float* out = (float*)d_out;

    char* ws = (char*)d_ws;
    unsigned short* wsw  = (unsigned short*)ws;                  // packed weights: 84 KB
    unsigned short* ws_u = (unsigned short*)(ws + (1 << 17));    // update gate, bf16: 32 MB
    unsigned short* ws_r = ws_u + (size_t)BD * CHN * HD * WD;    // reset gate,  bf16: 32 MB

    const int sh1 = (LIN_OFF1 + LROWS * LCOLS * CHPAD) * 2;     // 83264 B
    const int sh2 = (LIN_OFF2 + LROWS * LCOLS * CHPAD) * 2;     // 54592 B
    (void)hipFuncSetAttribute((const void*)convgru_pass1,
                              hipFuncAttributeMaxDynamicSharedMemorySize, sh1);
    (void)hipFuncSetAttribute((const void*)convgru_pass2,
                              hipFuncAttributeMaxDynamicSharedMemorySize, sh2);

    pack_weights<<<(NFRAG * 32 + 127) / 128, 128, 0, stream>>>(w_update, w_reset, w_out, wsw);

    dim3 grid(WD / TILE_W, HD / TILE_H, BD);
    convgru_pass1<<<grid, 256, sh1, stream>>>(x, prev, mask, wsw, b_update, b_reset, ws_u, ws_r);
    convgru_pass2<<<grid, 256, sh2, stream>>>(x, prev, mask, wsw, b_out, ws_u, ws_r, out);
}
